// GCNSelective_22428319219832
// MI455X (gfx1250) — matrix operand order repaired and checked by reading, not compiled, not run on hardware
//
#include <hip/hip_runtime.h>
#include <hip/hip_bf16.h>

typedef __attribute__((ext_vector_type(16))) _Float16 v16h;
typedef __attribute__((ext_vector_type(8)))  _Float16 v8h;
typedef __attribute__((ext_vector_type(8)))  float    v8f;

#define DH 128

// ---------------- small utility kernels ----------------

__global__ void k_fill1(float* p, int n) {
    int i = blockIdx.x * blockDim.x + threadIdx.x;
    if (i < n) p[i] = 1.0f;                     // self-loop contributes 1 to degree
}

__global__ void k_degcount(const int* __restrict__ dst, float* __restrict__ deg, int E) {
    int e = blockIdx.x * blockDim.x + threadIdx.x;
    if (e < E) atomicAdd(&deg[dst[e]], 1.0f);
}

__global__ void k_dinv(const float* __restrict__ deg, float* __restrict__ dinv, int n) {
    int i = blockIdx.x * blockDim.x + threadIdx.x;
    if (i < n) dinv[i] = rsqrtf(deg[i]);        // deg >= 1 always (self loops)
}

__global__ void k_cvt(const float* __restrict__ x, _Float16* __restrict__ h, int n) {
    int i = blockIdx.x * blockDim.x + threadIdx.x;
    if (i < n) h[i] = (_Float16)x[i];
}

// Pack W [128, dout] fp32 row-major into WMMA B-fragment order:
// wp[tile_n][kk][lane][16 halfs]; lane<16 -> K = kk*32 + 0..15, lane>=16 -> K = kk*32 + 16..31
__global__ void k_pack_w(const float* __restrict__ W, _Float16* __restrict__ wp,
                         int dout, int tn) {
    int t = blockIdx.x * blockDim.x + threadIdx.x;
    if (t >= tn * 4 * 32) return;
    int lane = t & 31, kk = (t >> 5) & 3, tile = t >> 7;
    int g = lane >> 4, n = (lane & 15) + tile * 16;
    _Float16* o = wp + (size_t)t * 16;
    for (int j = 0; j < 16; ++j) {
        int k = kk * 32 + g * 16 + j;
        o[j] = (n < dout) ? (_Float16)W[(size_t)k * dout + n] : (_Float16)0.0f;
    }
}

// ---------------- WMMA GEMM: hout[N, TN*16] = hin[N,128] @ Wpacked ----------------
// One wave per 16x16 output tile; wave w of the block owns column-tile w.
template<int TN>
__global__ void k_gemm(const _Float16* __restrict__ hin, const _Float16* __restrict__ wp,
                       _Float16* __restrict__ hout, int nrows) {
    const int wave = threadIdx.x >> 5;           // = tile_n
    const int lane = threadIdx.x & 31;
    const int g = lane >> 4, mr = lane & 15;
    const int row0 = blockIdx.x * 16;
    int row = row0 + mr;
    int srow = row < nrows ? row : nrows - 1;    // clamp; WMMA rows are independent
    const _Float16* a0 = hin + (size_t)srow * DH;

    v8f acc = {};
#pragma unroll
    for (int kk = 0; kk < 4; ++kk) {
        // A fragment: lane-group g holds K = kk*32 + g*16..+15 (matches packed B)
        v16h a = *(const v16h*)(a0 + kk * 32 + g * 16);
        v16h b = *(const v16h*)(wp + (((size_t)wave * 4 + kk) * 32 + lane) * 16);
        acc = __builtin_amdgcn_wmma_f32_16x16x32_f16(
            /*neg_a=*/false, a, /*neg_b=*/false, b,
            /*c_mod=*/(short)0, acc, /*reuse_a=*/false, /*reuse_b=*/false);
    }
    // D layout: VGPR i, lanes 0-15 -> M=i; lanes 16-31 -> M=i+8; N=lane%16
    const int col = wave * 16 + mr;
    _Float16* outp = hout + (size_t)(row0 + g * 8) * (TN * 16) + col;
    if (row0 + 16 <= nrows) {                    // fast path: full tile, straight-line stores
#pragma unroll
        for (int i = 0; i < 8; ++i) outp[(size_t)i * (TN * 16)] = (_Float16)acc[i];
    } else {                                     // only the last M-tile takes this path
#pragma unroll
        for (int i = 0; i < 8; ++i) {
            if (row0 + g * 8 + i < nrows) outp[(size_t)i * (TN * 16)] = (_Float16)acc[i];
        }
    }
}

// ---------------- edge scatter: agg[dst] += h[src] * dinv[src]*dinv[dst] ----------------

// 8 channels/thread for the 128-wide layers: one b128 f16 gather + 8 f32 atomics
__global__ void k_scatter8(const _Float16* __restrict__ h, const float* __restrict__ dinv,
                           const int* __restrict__ src, const int* __restrict__ dst,
                           float* __restrict__ agg, unsigned E) {
    unsigned t = blockIdx.x * blockDim.x + threadIdx.x;
    if (t >= E * 16u) return;                    // 128/8 = 16 groups per edge
    unsigned e = t >> 4;
    int cg = (int)(t & 15u) * 8;
    int s = src[e], d = dst[e];
    float norm = dinv[s] * dinv[d];
    v8h hv = *(const v8h*)(h + (size_t)s * DH + cg);
    float* ap = agg + (size_t)d * DH + cg;
#pragma unroll
    for (int j = 0; j < 8; ++j) atomicAdd(ap + j, (float)hv[j] * norm);
}

// generic 4 channels/thread (used by the 20-channel latent layer)
__global__ void k_scatter4(const _Float16* __restrict__ h, const float* __restrict__ dinv,
                           const int* __restrict__ src, const int* __restrict__ dst,
                           float* __restrict__ agg, unsigned E, int hstride, int nch) {
    unsigned gsz = (unsigned)(nch >> 2);
    unsigned t = blockIdx.x * blockDim.x + threadIdx.x;
    if (t >= E * gsz) return;
    unsigned e = t / gsz;
    int cg = (int)(t - e * gsz) * 4;
    int s = src[e], d = dst[e];
    float norm = dinv[s] * dinv[d];
    const _Float16* hp = h + (size_t)s * hstride + cg;
    float* ap = agg + (size_t)d * nch + cg;
#pragma unroll
    for (int j = 0; j < 4; ++j) atomicAdd(ap + j, (float)hp[j] * norm);
}

// ---------------- finalize: add self-loop msg + bias, optional relu, emit f16 and/or f32 ----------------
__global__ void k_finalize(const float* __restrict__ agg, const _Float16* __restrict__ hself,
                           const float* __restrict__ dinv, const float* __restrict__ bias,
                           _Float16* out16, float* out32,
                           int nrows, int nch, int hstride, int relu) {
    int t = blockIdx.x * blockDim.x + threadIdx.x;
    if (t >= nrows * nch) return;
    int r = t / nch, c = t - r * nch;
    float di = dinv[r];
    float v = agg[t] + (float)hself[(size_t)r * hstride + c] * di * di + bias[c];
    if (relu) v = fmaxf(v, 0.0f);
    if (out16) out16[t] = (_Float16)v;
    if (out32) out32[t] = v;
}

// ---------------- pooling + MLP epilogue ----------------

__global__ void k_offsets(const int* __restrict__ batch, int* __restrict__ offsets,
                          int N, int B) {
    int b = threadIdx.x;
    if (b > B) return;
    if (b == B) { offsets[B] = N; return; }
    int lo = 0, hi = N;                          // lower_bound(batch, b) — batch is sorted
    while (lo < hi) { int mid = (lo + hi) >> 1; if (batch[mid] < b) lo = mid + 1; else hi = mid; }
    offsets[b] = lo;
}

__global__ void k_pool(const float* __restrict__ h5, const int* __restrict__ offsets,
                       float* __restrict__ xg, int dlat) {
    int b = blockIdx.x / dlat, c = blockIdx.x % dlat;
    int lo = offsets[b], hi = offsets[b + 1];
    float s = 0.0f;
    for (int i = lo + threadIdx.x; i < hi; i += blockDim.x) s += h5[(size_t)i * dlat + c];
    __shared__ float red[256];
    red[threadIdx.x] = s; __syncthreads();
    for (int st = blockDim.x >> 1; st > 0; st >>= 1) {
        if (threadIdx.x < st) red[threadIdx.x] += red[threadIdx.x + st];
        __syncthreads();
    }
    if (threadIdx.x == 0) xg[b * dlat + c] = red[0] / (float)(hi - lo);
}

__global__ void k_mlp(const float* __restrict__ h5, const float* __restrict__ xg,
                      const int* __restrict__ base, const int* __restrict__ offsets,
                      const float* __restrict__ w1, const float* __restrict__ b1,
                      const float* __restrict__ w2, const float* __restrict__ b2,
                      float* __restrict__ out, int L, int dlat) {
    int b = blockIdx.x, tid = threadIdx.x;       // blockDim = 128
    __shared__ float z[240];
    __shared__ float red[128];
    int zs = L * dlat;                           // 200
    for (int i = tid; i < zs; i += blockDim.x) { // x_s (selective pool, base==0 -> zeros)
        int l = i / dlat, c = i - l * dlat;
        int bs = base[b * L + l];
        float v = 0.0f;
        if (bs != 0) { int gi = offsets[b] + bs; v = h5[(size_t)gi * dlat + c]; }
        z[i] = v;
    }
    for (int i = tid; i < dlat; i += blockDim.x) z[zs + i] = xg[b * dlat + i];  // x_g
    __syncthreads();
    int zt = zs + dlat;                          // 220
    float acc = b1[tid];
    for (int i = 0; i < zt; ++i) acc += z[i] * w1[(size_t)i * 128 + tid];
    acc = fmaxf(acc, 0.0f);
    red[tid] = acc * w2[tid];
    __syncthreads();
    for (int st = 64; st > 0; st >>= 1) { if (tid < st) red[tid] += red[tid + st]; __syncthreads(); }
    if (tid == 0) out[b] = red[0] + b2[0];
}

// ---------------- host launcher ----------------

extern "C" void kernel_launch(void* const* d_in, const int* in_sizes, int n_in,
                              void* d_out, int out_size, void* d_ws, size_t ws_size,
                              hipStream_t stream) {
    const float* x     = (const float*)d_in[0];
    const int*   ei    = (const int*)d_in[1];
    const int*   batch = (const int*)d_in[2];
    const int*   base  = (const int*)d_in[3];
    const float* W[5]  = { (const float*)d_in[4], (const float*)d_in[6], (const float*)d_in[8],
                           (const float*)d_in[10], (const float*)d_in[12] };
    const float* bi[5] = { (const float*)d_in[5], (const float*)d_in[7], (const float*)d_in[9],
                           (const float*)d_in[11], (const float*)d_in[13] };
    const float* l1w = (const float*)d_in[14];
    const float* l1b = (const float*)d_in[15];
    const float* l2w = (const float*)d_in[16];
    const float* l2b = (const float*)d_in[17];
    float* out = (float*)d_out;

    const int N    = in_sizes[0] / DH;
    const int E    = in_sizes[1] / 2;
    const int L    = 10;
    const int B    = in_sizes[3] / L;
    const int DLAT = in_sizes[13];               // 20

    const int* src = ei;
    const int* dst = ei + E;

    char* p = (char*)d_ws;
    auto alloc = [&](size_t bytes) -> void* {
        void* r = (void*)p; p += (bytes + 255) & ~(size_t)255; return r;
    };
    float*    deg     = (float*)alloc((size_t)N * 4);
    float*    dinv    = (float*)alloc((size_t)N * 4);
    _Float16* hin     = (_Float16*)alloc((size_t)N * DH * 2);   // layer input, f16
    _Float16* hg      = (_Float16*)alloc((size_t)N * DH * 2);   // GEMM output, f16
    float*    agg     = (float*)alloc((size_t)N * DH * 4);      // fp32 scatter accumulator
    float*    h5      = (float*)alloc((size_t)N * DLAT * 4);    // final latent, fp32
    _Float16* wpack   = (_Float16*)alloc((size_t)8 * 4 * 32 * 16 * 2);
    int*      offsets = (int*)alloc((size_t)(B + 1) * 4);
    float*    xg      = (float*)alloc((size_t)B * DLAT * 4);
    (void)ws_size; (void)n_in; (void)out_size;

    const int TB = 256;
    auto cdiv = [](long long a, long long b) { return (int)((a + b - 1) / b); };

    // degree (with self-loop) -> dinv; x -> f16
    k_fill1   <<<cdiv(N, TB), TB, 0, stream>>>(deg, N);
    k_degcount<<<cdiv(E, TB), TB, 0, stream>>>(dst, deg, E);
    k_dinv    <<<cdiv(N, TB), TB, 0, stream>>>(deg, dinv, N);
    k_cvt     <<<cdiv((long long)N * DH, TB), TB, 0, stream>>>(x, hin, N * DH);

    const int mtiles = cdiv(N, 16);

    // layers 1-4: 128 -> 128, relu
    for (int l = 0; l < 4; ++l) {
        k_pack_w<<<cdiv(8 * 4 * 32, TB), TB, 0, stream>>>(W[l], wpack, DH, 8);
        k_gemm<8><<<mtiles, 8 * 32, 0, stream>>>(hin, wpack, hg, N);
        hipMemsetAsync(agg, 0, (size_t)N * DH * 4, stream);
        k_scatter8<<<cdiv((long long)E * 16, TB), TB, 0, stream>>>(
            hg, dinv, src, dst, agg, (unsigned)E);
        k_finalize<<<cdiv((long long)N * DH, TB), TB, 0, stream>>>(
            agg, hg, dinv, bi[l], hin, nullptr, N, DH, DH, 1);
    }

    // layer 5: 128 -> 20 (cols padded to 32 for WMMA), no relu, fp32 output
    k_pack_w<<<cdiv(2 * 4 * 32, TB), TB, 0, stream>>>(W[4], wpack, DLAT, 2);
    k_gemm<2><<<mtiles, 2 * 32, 0, stream>>>(hin, wpack, hg, N);
    hipMemsetAsync(agg, 0, (size_t)N * DLAT * 4, stream);
    k_scatter4<<<cdiv((long long)E * (DLAT / 4), TB), TB, 0, stream>>>(
        hg, dinv, src, dst, agg, (unsigned)E, 32, DLAT);
    k_finalize<<<cdiv((long long)N * DLAT, TB), TB, 0, stream>>>(
        agg, hg, dinv, bi[4], nullptr, h5, N, DLAT, 32, 0);

    // pooling + MLP head
    k_offsets<<<1, B + 1, 0, stream>>>(batch, offsets, N, B);
    k_pool<<<B * DLAT, 256, 0, stream>>>(h5, offsets, xg, DLAT);
    k_mlp<<<B, 128, 0, stream>>>(h5, xg, base, offsets, l1w, l1b, l2w, l2b, out, L, DLAT);
}
